// cordsnet_63367947485571
// MI455X (gfx1250) — matrix operand order repaired and checked by reading, not compiled, not run on hardware
//
#include <hip/hip_runtime.h>
#include <hip/hip_bf16.h>
#include <math.h>

// ---------------------------------------------------------------------------
// Types for CDNA5 WMMA (wave32): v_wmma_f32_16x16x32_bf16
// ---------------------------------------------------------------------------
typedef __attribute__((ext_vector_type(16))) __bf16 v16bf;
typedef __attribute__((ext_vector_type(8)))  __bf16 bf16x8;
typedef __attribute__((ext_vector_type(8)))  float  v8f;
typedef __attribute__((ext_vector_type(4)))  float  v4f;

// ---------------------------------------------------------------------------
// Load one 16-float channel chunk (optionally plus a second source, each with
// optional ReLU) and convert to this lane's bf16 B-fragment chunk.
// B k order used here (contiguous, same as A): lane<16 -> col N=lane, K=0..15 ; lane>=16 -> K=16..31
// Pointers are ALWAYS valid (padding lanes are redirected to a zero page).
// ---------------------------------------------------------------------------
__device__ __forceinline__ v16bf load_act_frag(const float* pa, const float* pb,
                                               int reluA, int reluB) {
  v4f a0 = ((const v4f*)pa)[0];
  v4f a1 = ((const v4f*)pa)[1];
  v4f a2 = ((const v4f*)pa)[2];
  v4f a3 = ((const v4f*)pa)[3];
  float fa[16];
#pragma unroll
  for (int i = 0; i < 4; ++i) {
    fa[i] = a0[i]; fa[4 + i] = a1[i]; fa[8 + i] = a2[i]; fa[12 + i] = a3[i];
  }
  if (reluA) {
#pragma unroll
    for (int i = 0; i < 16; ++i) fa[i] = fmaxf(fa[i], 0.0f);
  }
  if (pb) {   // uniform (scalar) condition
    v4f b0 = ((const v4f*)pb)[0];
    v4f b1 = ((const v4f*)pb)[1];
    v4f b2 = ((const v4f*)pb)[2];
    v4f b3 = ((const v4f*)pb)[3];
    float fb[16];
#pragma unroll
    for (int i = 0; i < 4; ++i) {
      fb[i] = b0[i]; fb[4 + i] = b1[i]; fb[8 + i] = b2[i]; fb[12 + i] = b3[i];
    }
#pragma unroll
    for (int i = 0; i < 16; ++i)
      fa[i] += reluB ? fmaxf(fb[i], 0.0f) : fb[i];
  }
  v16bf r;
#pragma unroll
  for (int i = 0; i < 16; ++i) r[i] = (__bf16)fa[i];
  return r;
}

// A-fragment, contiguous k order as in B: lane<16 -> row M=lane&15, K=0..15;
// lane>=16 -> K=16..31. wrow points at W[tap][coBase+m][kc*32 + half*16].
__device__ __forceinline__ v16bf load_w_frag(const __bf16* wrow) {
  bf16x8 lo = *(const bf16x8*)(wrow);
  bf16x8 hi = *(const bf16x8*)(wrow + 8);
  v16bf a;
#pragma unroll
  for (int i = 0; i < 8; ++i) { a[i] = lo[i]; a[8 + i] = hi[i]; }
  return a;
}

// ---------------------------------------------------------------------------
// Generic dual-conv implicit-GEMM WMMA kernel (NHWC f32 activations, tap-major
// bf16 weights, f32 accumulators).
//   pre = conv3x3_s1_p1(rin, W1) + convK_sS(sumReLU(sA,sB), W2) + bias
//   epilogue: if alphaP: out = (1-a)*rin + a*relu(pre)   (integrator update)
//             else:      out = pre                        (plain conv)
// Register blocking: each wave computes 64(cout) x 16(pixel) via 4 WMMA
// accumulators sharing one B fragment (4x reuse of activation loads + cvt).
// Block = 128 thr = 4 waves on 4 consecutive pixel tiles;
// grid = (ceil(N/64), Cout/64).
// ---------------------------------------------------------------------------
__global__ __launch_bounds__(128) void k_dual_conv(
    const float* __restrict__ rin, const __bf16* __restrict__ W1,
    const float* __restrict__ sA, const float* __restrict__ sB,
    int reluA, int reluB,
    const __bf16* __restrict__ W2, int k2, int s2, int C2, int Hs, int Ws,
    const float* __restrict__ biasT, const float* __restrict__ alphaP,
    const float* __restrict__ zpad,
    float* __restrict__ outp, int Bn, int H, int W, int Cout) {
  const int tid  = threadIdx.x;
  const int lane = tid & 31;
  const int wave = tid >> 5;
  const int m    = lane & 15;
  const int half = lane >> 4;
  const int coBase = blockIdx.y * 64;
  const int N = Bn * H * W;
  const int pix = (blockIdx.x * 4 + wave) * 16 + m;
  const bool pvalid = pix < N;
  const int pp = pvalid ? pix : 0;
  const int ob = pp / (H * W);
  const int rr = pp - ob * H * W;
  const int oy = rr / W;
  const int ox = rr - oy * W;

  v8f acc[4];
#pragma unroll
  for (int t = 0; t < 4; ++t)
    acc[t] = (v8f){0.f, 0.f, 0.f, 0.f, 0.f, 0.f, 0.f, 0.f};

  // ---- conv1: 3x3 stride-1 pad-1 on rin (Cin == Cout) --------------------
  if (W1) {
    const int KC = Cout >> 5;
    for (int ky = 0; ky < 3; ++ky) {
      for (int kx = 0; kx < 3; ++kx) {
        const int iy = oy + ky - 1;
        const int ix = ox + kx - 1;
        const bool inb = pvalid && iy >= 0 && iy < H && ix >= 0 && ix < W;
        const float* ap =
            inb ? rin + (((size_t)ob * H + iy) * W + ix) * Cout + half * 16
                : zpad;
        const size_t stepA = inb ? 32 : 0;
        const __bf16* w0 =
            W1 + ((size_t)((ky * 3 + kx) * Cout + coBase + m)) * Cout + half * 16;
        for (int kc = 0; kc < KC; ++kc) {
          v16bf bf = load_act_frag(ap, nullptr, 0, 0);
          ap += stepA;
#pragma unroll
          for (int t = 0; t < 4; ++t) {
            v16bf af = load_w_frag(w0 + (size_t)t * 16 * Cout + kc * 32);
            acc[t] = __builtin_amdgcn_wmma_f32_16x16x32_bf16(
                false, af, false, bf, (short)0, acc[t], false, false);
          }
        }
      }
    }
  }

  // ---- conv2: k2 x k2, stride s2, input = reluA(sA) + reluB(sB) ----------
  if (W2) {
    const int KC2 = C2 >> 5;
    const int pad = (k2 == 3) ? 1 : 0;
    for (int ky = 0; ky < k2; ++ky) {
      for (int kx = 0; kx < k2; ++kx) {
        const int iy = oy * s2 + ky - pad;
        const int ix = ox * s2 + kx - pad;
        const bool inb = pvalid && iy >= 0 && iy < Hs && ix >= 0 && ix < Ws;
        const size_t so =
            (((size_t)ob * Hs + iy) * Ws + ix) * (size_t)C2 + half * 16;
        const float* ap = inb ? sA + so : zpad;
        const float* bp = sB ? (inb ? sB + so : zpad) : nullptr;
        const size_t stepA = inb ? 32 : 0;
        const __bf16* w0 =
            W2 + ((size_t)((ky * k2 + kx) * Cout + coBase + m)) * C2 + half * 16;
        for (int kc = 0; kc < KC2; ++kc) {
          v16bf bf = load_act_frag(ap, bp, reluA, reluB);
          ap += stepA;
          if (bp) bp += stepA;
#pragma unroll
          for (int t = 0; t < 4; ++t) {
            v16bf af = load_w_frag(w0 + (size_t)t * 16 * C2 + kc * 32);
            acc[t] = __builtin_amdgcn_wmma_f32_16x16x32_bf16(
                false, af, false, bf, (short)0, acc[t], false, false);
          }
        }
      }
    }
  }

  // ---- epilogue: C/D layout -> lane L: M = v + 8*(L>=16), N = L&15 -------
  if (pvalid) {
    const size_t pbase = (((size_t)ob * H + oy) * W + ox) * (size_t)Cout;
    const size_t bbase = ((size_t)(oy * W + ox)) * (size_t)Cout;
    const float al = alphaP ? *alphaP : 0.0f;
#pragma unroll
    for (int t = 0; t < 4; ++t) {
      const int co0 = coBase + t * 16 + half * 8;
      const size_t o = pbase + co0;
      float bs[8];
#pragma unroll
      for (int v = 0; v < 8; ++v) bs[v] = 0.0f;
      if (biasT) {
        const float* bp = biasT + bbase + co0;
        v4f b0 = ((const v4f*)bp)[0];
        v4f b1 = ((const v4f*)bp)[1];
#pragma unroll
        for (int i = 0; i < 4; ++i) { bs[i] = b0[i]; bs[4 + i] = b1[i]; }
      }
      float outv[8];
      if (alphaP) {
        const float* rp = rin + o;
        v4f r0 = ((const v4f*)rp)[0];
        v4f r1 = ((const v4f*)rp)[1];
        float ro[8];
#pragma unroll
        for (int i = 0; i < 4; ++i) { ro[i] = r0[i]; ro[4 + i] = r1[i]; }
#pragma unroll
        for (int v = 0; v < 8; ++v) {
          float pre = acc[t][v] + bs[v];
          outv[v] = (1.0f - al) * ro[v] + al * fmaxf(pre, 0.0f);
        }
      } else {
#pragma unroll
        for (int v = 0; v < 8; ++v) outv[v] = acc[t][v] + bs[v];
      }
      v4f s0, s1;
#pragma unroll
      for (int i = 0; i < 4; ++i) { s0[i] = outv[i]; s1[i] = outv[4 + i]; }
      ((v4f*)(outp + o))[0] = s0;
      ((v4f*)(outp + o))[1] = s1;
    }
  }
}

// ---------------------------------------------------------------------------
// Weight-norm + repack to tap-major bf16: v[Cout][Cin][K][K], g[Cout]
//   -> out[(k)*Cout + co][Cin] bf16, w = g*v/||v_co||
// ---------------------------------------------------------------------------
__global__ __launch_bounds__(256) void k_wn_conv_bf16(
    const float* __restrict__ v, const float* __restrict__ g,
    __bf16* __restrict__ out, int Cout, int Cin, int KK) {
  __shared__ float sh[256];
  const int co = blockIdx.x;
  const int tid = threadIdx.x;
  const int n = Cin * KK;
  const float* row = v + (size_t)co * n;
  float s = 0.0f;
  for (int i = tid; i < n; i += 256) { float t = row[i]; s += t * t; }
  sh[tid] = s;
  __syncthreads();
  for (int w = 128; w > 0; w >>= 1) {
    if (tid < w) sh[tid] += sh[tid + w];
    __syncthreads();
  }
  const float inv = g[co] / sqrtf(sh[0]);
  for (int i = tid; i < n; i += 256) {
    const int ci = i / KK;
    const int t = i - ci * KK;
    out[((size_t)t * Cout + co) * Cin + ci] = (__bf16)(row[i] * inv);
  }
}

// Weight-norm, fp32 output, same layout (for input stem & FC).
__global__ __launch_bounds__(256) void k_wn_plain_f32(
    const float* __restrict__ v, const float* __restrict__ g,
    float* __restrict__ out, int rowlen) {
  __shared__ float sh[256];
  const int r = blockIdx.x;
  const int tid = threadIdx.x;
  const float* row = v + (size_t)r * rowlen;
  float s = 0.0f;
  for (int i = tid; i < rowlen; i += 256) { float t = row[i]; s += t * t; }
  sh[tid] = s;
  __syncthreads();
  for (int w = 128; w > 0; w >>= 1) {
    if (tid < w) sh[tid] += sh[tid + w];
    __syncthreads();
  }
  const float inv = g[r] / sqrtf(sh[0]);
  for (int i = tid; i < rowlen; i += 256) out[(size_t)r * rowlen + i] = row[i] * inv;
}

// bias [C][H][W] -> [H*W][C]
__global__ void k_bias_t(const float* __restrict__ in, float* __restrict__ out,
                         int C, int HW) {
  const int i = blockIdx.x * 256 + threadIdx.x;
  if (i >= C * HW) return;
  const int c = i / HW;
  const int p = i - c * HW;
  out[(size_t)p * C + c] = in[i];
}

__global__ void k_fill0(float* __restrict__ p, int n) {
  const int i = blockIdx.x * 256 + threadIdx.x;
  if (i < n) p[i] = 0.0f;
}

// Input stem conv: [2,3,224,224] NCHW * Wn[64][3][7][7] stride2 pad3 -> NHWC [2,112,112,64]
__global__ void k_conv_in(const float* __restrict__ in,
                          const float* __restrict__ Wn,
                          float* __restrict__ out) {
  const int i = blockIdx.x * 256 + threadIdx.x;
  if (i >= 2 * 112 * 112 * 64) return;
  const int co = i & 63;
  int p = i >> 6;
  const int xo = p % 112; p /= 112;
  const int yo = p % 112;
  const int b = p / 112;
  float s = 0.0f;
  for (int ci = 0; ci < 3; ++ci)
    for (int ky = 0; ky < 7; ++ky) {
      const int iy = yo * 2 + ky - 3;
      if (iy < 0 || iy >= 224) continue;
      for (int kx = 0; kx < 7; ++kx) {
        const int ix = xo * 2 + kx - 3;
        if (ix < 0 || ix >= 224) continue;
        s += in[((size_t)(b * 3 + ci) * 224 + iy) * 224 + ix] *
             Wn[((size_t)(co * 3 + ci) * 7 + ky) * 7 + kx];
      }
    }
  out[((size_t)(b * 112 + yo) * 112 + xo) * 64 + co] = s;
}

// AvgPool 3x3 s2 p1 (count_include_pad): NHWC [2,112,112,64] -> [2,56,56,64]
__global__ void k_avgpool(const float* __restrict__ in, float* __restrict__ out) {
  const int i = blockIdx.x * 256 + threadIdx.x;
  if (i >= 2 * 56 * 56 * 64) return;
  const int c = i & 63;
  int p = i >> 6;
  const int xo = p % 56; p /= 56;
  const int yo = p % 56;
  const int b = p / 56;
  float s = 0.0f;
  for (int ky = 0; ky < 3; ++ky) {
    const int iy = yo * 2 + ky - 1;
    if (iy < 0 || iy >= 112) continue;
    for (int kx = 0; kx < 3; ++kx) {
      const int ix = xo * 2 + kx - 1;
      if (ix < 0 || ix >= 112) continue;
      s += in[((size_t)(b * 112 + iy) * 112 + ix) * 64 + c];
    }
  }
  out[((size_t)(b * 56 + yo) * 56 + xo) * 64 + c] = s * (1.0f / 9.0f);
}

// pooled[b][c] = mean over 7x7 of relu(rs7)+relu(rs6); NHWC [2,7,7,512]
__global__ void k_pool(const float* __restrict__ r6, const float* __restrict__ r7,
                       float* __restrict__ pooled) {
  const int i = blockIdx.x * 256 + threadIdx.x;
  if (i >= 2 * 512) return;
  const int c = i & 511;
  const int b = i >> 9;
  float s = 0.0f;
  for (int p = 0; p < 49; ++p) {
    const size_t o = ((size_t)b * 49 + p) * 512 + c;
    s += fmaxf(r7[o], 0.0f) + fmaxf(r6[o], 0.0f);
  }
  pooled[i] = s * (1.0f / 49.0f);
}

// out[b][o] = pooled[b] . Wfc[o] + fcb[o]
__global__ void k_fc(const float* __restrict__ pooled, const float* __restrict__ Wfc,
                     const float* __restrict__ fcb, float* __restrict__ out) {
  const int i = blockIdx.x * 256 + threadIdx.x;
  if (i >= 2 * 1000) return;
  const int o = i % 1000;
  const int b = i / 1000;
  float s = fcb[o];
  const float* pr = pooled + b * 512;
  const float* wr = Wfc + (size_t)o * 512;
  for (int c = 0; c < 512; ++c) s += pr[c] * wr[c];
  out[i] = s;
}

// ---------------------------------------------------------------------------
// Host orchestration
// ---------------------------------------------------------------------------
extern "C" void kernel_launch(void* const* d_in, const int* in_sizes, int n_in,
                              void* d_out, int out_size, void* d_ws, size_t ws_size,
                              hipStream_t stream) {
  (void)in_sizes; (void)n_in; (void)out_size; (void)ws_size;
  static const int CHh[9]  = {64, 64, 64, 128, 128, 256, 256, 512, 512};
  static const int SZh[8]  = {56, 56, 28, 28, 14, 14, 7, 7};
  static const int SKIP[4] = {64, 128, 256, 512};

  // ---- inputs (setup_inputs dict order) ----
  const float* inputs     = (const float*)d_in[0];
  const float* alpha      = (const float*)d_in[1];
  const float* inp_conv_v = (const float*)d_in[3];
  const float* inp_conv_g = (const float*)d_in[4];
  const float* inp_skip_v = (const float*)d_in[5];
  const float* inp_skip_g = (const float*)d_in[6];
  const float* area_conv_v[8]; const float* area_conv_g[8];
  const float* conv_bias[8];   const float* area_area_v[8]; const float* area_area_g[8];
  for (int j = 0; j < 8; ++j) {
    area_conv_v[j] = (const float*)d_in[7 + j];
    area_conv_g[j] = (const float*)d_in[15 + j];
    conv_bias[j]   = (const float*)d_in[23 + j];
    area_area_v[j] = (const float*)d_in[31 + j];
    area_area_g[j] = (const float*)d_in[39 + j];
  }
  const float* skip_v[3]; const float* skip_g[3];
  for (int k = 0; k < 3; ++k) {
    skip_v[k] = (const float*)d_in[47 + k];
    skip_g[k] = (const float*)d_in[50 + k];
  }
  const float* fc_v = (const float*)d_in[53];
  const float* fc_g = (const float*)d_in[54];
  const float* fc_b = (const float*)d_in[55];
  float* out = (float*)d_out;

  // ---- workspace carve (deterministic bump allocator) ----
  char* base = (char*)d_ws;
  size_t off = 0;
  auto alloc = [&](size_t bytes) -> char* {
    off = (off + 255) & ~(size_t)255;
    char* p = base + off;
    off += bytes;
    return p;
  };
  float*  zpad   = (float*)alloc(64 * 4);       // zero page for padding lanes
  float*  W_in_n = (float*)alloc((size_t)64 * 147 * 4);
  float*  Wfc_n  = (float*)alloc((size_t)1000 * 512 * 4);
  __bf16* Wis    = (__bf16*)alloc((size_t)9 * 64 * 64 * 2);
  __bf16* Wac[8]; __bf16* Waa[8];
  for (int j = 0; j < 8; ++j) {
    Wac[j] = (__bf16*)alloc((size_t)9 * CHh[j + 1] * CHh[j + 1] * 2);
    Waa[j] = (__bf16*)alloc((size_t)9 * CHh[j + 1] * CHh[j] * 2);
  }
  __bf16* Wsk[3];
  for (int k = 0; k < 3; ++k) Wsk[k] = (__bf16*)alloc((size_t)SKIP[k + 1] * SKIP[k] * 2);
  float* biasT[8];
  for (int j = 0; j < 8; ++j)
    biasT[j] = (float*)alloc((size_t)SZh[j] * SZh[j] * CHh[j + 1] * 4);
  float* tmp112 = (float*)alloc((size_t)2 * 112 * 112 * 64 * 4);
  float* inp    = (float*)alloc((size_t)2 * 56 * 56 * 64 * 4);
  float* inps   = (float*)alloc((size_t)2 * 56 * 56 * 64 * 4);
  float* rsA[8]; float* rsB[8]; int rsN[8];
  for (int j = 0; j < 8; ++j) {
    rsN[j] = 2 * SZh[j] * SZh[j] * CHh[j + 1];
    rsA[j] = (float*)alloc((size_t)rsN[j] * 4);
    rsB[j] = (float*)alloc((size_t)rsN[j] * 4);
  }
  float* skipbuf[3];
  for (int k = 0; k < 3; ++k)
    skipbuf[k] = (float*)alloc((size_t)2 * SZh[2 * k + 2] * SZh[2 * k + 2] * SKIP[k + 1] * 4);
  float* pooled = (float*)alloc((size_t)2 * 512 * 4);

  auto cdiv = [](int a, int b) { return (a + b - 1) / b; };

  // ---- zero page + weight-norm + repack ----
  hipLaunchKernelGGL(k_fill0, dim3(1), dim3(256), 0, stream, zpad, 64);
  hipLaunchKernelGGL(k_wn_plain_f32, dim3(64), dim3(256), 0, stream,
                     inp_conv_v, inp_conv_g, W_in_n, 147);
  hipLaunchKernelGGL(k_wn_plain_f32, dim3(1000), dim3(256), 0, stream,
                     fc_v, fc_g, Wfc_n, 512);
  hipLaunchKernelGGL(k_wn_conv_bf16, dim3(64), dim3(256), 0, stream,
                     inp_skip_v, inp_skip_g, Wis, 64, 64, 9);
  for (int j = 0; j < 8; ++j) {
    hipLaunchKernelGGL(k_wn_conv_bf16, dim3(CHh[j + 1]), dim3(256), 0, stream,
                       area_conv_v[j], area_conv_g[j], Wac[j], CHh[j + 1], CHh[j + 1], 9);
    hipLaunchKernelGGL(k_wn_conv_bf16, dim3(CHh[j + 1]), dim3(256), 0, stream,
                       area_area_v[j], area_area_g[j], Waa[j], CHh[j + 1], CHh[j], 9);
    const int nb = CHh[j + 1] * SZh[j] * SZh[j];
    hipLaunchKernelGGL(k_bias_t, dim3(cdiv(nb, 256)), dim3(256), 0, stream,
                       conv_bias[j], biasT[j], CHh[j + 1], SZh[j] * SZh[j]);
  }
  for (int k = 0; k < 3; ++k)
    hipLaunchKernelGGL(k_wn_conv_bf16, dim3(SKIP[k + 1]), dim3(256), 0, stream,
                       skip_v[k], skip_g[k], Wsk[k], SKIP[k + 1], SKIP[k], 1);

  // ---- input drive (constant across timesteps) ----
  hipLaunchKernelGGL(k_conv_in, dim3(cdiv(2 * 112 * 112 * 64, 256)), dim3(256), 0,
                     stream, inputs, W_in_n, tmp112);
  hipLaunchKernelGGL(k_avgpool, dim3(cdiv(2 * 56 * 56 * 64, 256)), dim3(256), 0,
                     stream, tmp112, inp);

  auto dual = [&](const float* rin, const __bf16* W1, const float* sA,
                  const float* sB, int reluA, int reluB, const __bf16* W2, int k2,
                  int s2, int C2, int Hs, int Ws, const float* bT,
                  const float* aP, float* op, int H, int W, int Cout) {
    const int N = 2 * H * W;
    dim3 grid(cdiv(N, 64), Cout / 64);
    hipLaunchKernelGGL(k_dual_conv, grid, dim3(128), 0, stream, rin, W1, sA, sB,
                       reluA, reluB, W2, k2, s2, C2, Hs, Ws, bT, aP, zpad, op, 2,
                       H, W, Cout);
  };

  // inps = conv3x3_s1_p1(inp, W_is)
  dual(inp, Wis, nullptr, nullptr, 0, 0, nullptr, 3, 1, 64, 56, 56,
       nullptr, nullptr, inps, 56, 56, 64);

  // ---- zero-init recurrent state (cur buffers) ----
  for (int j = 0; j < 8; ++j)
    hipLaunchKernelGGL(k_fill0, dim3(cdiv(rsN[j], 256)), dim3(256), 0, stream,
                       rsA[j], rsN[j]);

  float* cur[8]; float* nxt[8];
  for (int j = 0; j < 8; ++j) { cur[j] = rsA[j]; nxt[j] = rsB[j]; }

  // ---- 100 warmup (zero drive) + 100 driven timesteps ----
  for (int t = 0; t < 200; ++t) {
    const bool driven = (t >= 100);
    // skip convs: 1x1 stride-2 on relu(r[1]), relu(r[3]), relu(r[5])
    for (int k = 0; k < 3; ++k) {
      const int js = 2 * k + 1;
      dual(nullptr, nullptr, cur[js], nullptr, 1, 0, Wsk[k], 1, 2, SKIP[k],
           SZh[js], SZh[js], nullptr, nullptr, skipbuf[k], SZh[2 * k + 2],
           SZh[2 * k + 2], SKIP[k + 1]);
    }
    // area 0: ain = inp (driven) or 0 (warmup)
    dual(cur[0], Wac[0], driven ? inp : nullptr, nullptr, 0, 0,
         driven ? Waa[0] : nullptr, 3, 1, 64, 56, 56, biasT[0], alpha, nxt[0],
         56, 56, 64);
    // area 1: ain = relu(r0) + inps (driven only)
    dual(cur[1], Wac[1], cur[0], driven ? inps : nullptr, 1, 0, Waa[1], 3, 1, 64,
         56, 56, biasT[1], alpha, nxt[1], 56, 56, 64);
    for (int j = 2; j < 8; ++j) {
      if ((j & 1) == 0) {
        // even: ain = relu(r[j-1]) + relu(r[j-2]), stride 2
        dual(cur[j], Wac[j], cur[j - 1], cur[j - 2], 1, 1, Waa[j], 3, 2, CHh[j],
             SZh[j - 1], SZh[j - 1], biasT[j], alpha, nxt[j], SZh[j], SZh[j],
             CHh[j + 1]);
      } else {
        // odd: ain = relu(r[j-1]) + skip[(j-3)/2], stride 1
        dual(cur[j], Wac[j], cur[j - 1], skipbuf[(j - 3) / 2], 1, 0, Waa[j], 3, 1,
             CHh[j], SZh[j], SZh[j], biasT[j], alpha, nxt[j], SZh[j], SZh[j],
             CHh[j + 1]);
      }
    }
    for (int j = 0; j < 8; ++j) { float* tpm = cur[j]; cur[j] = nxt[j]; nxt[j] = tpm; }
  }

  // ---- head: pooling + FC ----
  hipLaunchKernelGGL(k_pool, dim3(cdiv(2 * 512, 256)), dim3(256), 0, stream,
                     cur[6], cur[7], pooled);
  hipLaunchKernelGGL(k_fc, dim3(cdiv(2 * 1000, 256)), dim3(256), 0, stream,
                     pooled, Wfc_n, fc_b, out);
}
